// ParallelRetention_plain_46643344835369
// MI455X (gfx1250) — matrix operand order repaired and checked by reading, not compiled, not run on hardware
//
#include <hip/hip_runtime.h>

#define B_ 2
#define H_ 8
#define S_ 2048
#define D_ 128
#define EPS_ 1e-6f

typedef __attribute__((ext_vector_type(16))) __bf16 v16bf;
typedef __attribute__((ext_vector_type(8)))  float  v8f;
typedef __attribute__((ext_vector_type(4)))  unsigned int u32x4;

union BF16x16 {
  v16bf v;
  unsigned short s[16];
  u32x4 q[2];
};

// round-to-nearest-even f32 -> bf16 (bit pattern)
__device__ __forceinline__ unsigned short f2bf(float f) {
  union { float f; unsigned u; } x; x.f = f;
  unsigned r = x.u + 0x7FFFu + ((x.u >> 16) & 1u);
  return (unsigned short)(r >> 16);
}

// low 32 bits of a generic LDS pointer == byte offset within the wave's LDS
__device__ __forceinline__ unsigned lds_off(const void* p) {
  return (unsigned)(unsigned long long)p;
}

// CDNA5 async global->LDS copy, 16 bytes per lane (GVS addressing: s64 + v32)
__device__ __forceinline__ void async_b128(unsigned lds, unsigned voff, const void* sbase) {
  asm volatile("global_load_async_to_lds_b128 %0, %1, %2 offset:0"
               :: "v"(lds), "v"(voff), "s"(sbase) : "memory");
}
// same, non-temporal (streaming data: touched exactly once -> don't churn L2)
__device__ __forceinline__ void async_b128_nt(unsigned lds, unsigned voff, const void* sbase) {
  asm volatile("global_load_async_to_lds_b128 %0, %1, %2 offset:0 th:TH_LOAD_NT"
               :: "v"(lds), "v"(voff), "s"(sbase) : "memory");
}
__device__ __forceinline__ void wait_async0() {
  asm volatile("s_wait_asynccnt 0x0" ::: "memory");
}

// ---------- prepass 1: rs[h*S+i] = rsqrt(sum_j omask[h,i,j]), nan->0 ----------
__global__ __launch_bounds__(256) void retention_rowsum(const float* __restrict__ omask,
                                                        float* __restrict__ rs) {
  int row = blockIdx.x;  // H_*S_ rows
  const float* p = omask + (size_t)row * S_;
  float s = 0.f;
  for (int j = threadIdx.x; j < S_; j += 256) s += p[j];
  __shared__ float red[256];
  red[threadIdx.x] = s;
  __syncthreads();
  for (int off = 128; off > 0; off >>= 1) {
    if ((int)threadIdx.x < off) red[threadIdx.x] += red[threadIdx.x + off];
    __syncthreads();
  }
  if (threadIdx.x == 0) {
    float sum = red[0];
    rs[row] = (sum > 0.f) ? rsqrtf(sum) : 0.f;
  }
}

// ---------- prepass 2: k (f32 [b,h,s,d]) -> bf16, packed copies ----------
__global__ __launch_bounds__(256) void convert_k(const float* __restrict__ k,
                                                 unsigned short* __restrict__ kb) {
  size_t i = ((size_t)blockIdx.x * 256 + threadIdx.x) * 8;
  unsigned o[4];
  #pragma unroll
  for (int t = 0; t < 4; ++t) {
    unsigned lo = f2bf(k[i + 2 * t]);
    unsigned hiw = f2bf(k[i + 2 * t + 1]);
    o[t] = lo | (hiw << 16);
  }
  *(u32x4*)(kb + i) = *(u32x4*)o;
}

// ---------- prepass 3: v (f32 [b,h,s,d]) -> bf16 transposed [b,h,d,s] ----------
__global__ __launch_bounds__(256) void transpose_v(const float* __restrict__ v,
                                                   unsigned short* __restrict__ vT) {
  __shared__ unsigned short tile[32][33];
  const int bh = blockIdx.z;
  const int s0 = blockIdx.x * 32;
  const int d0 = blockIdx.y * 32;
  const int tx = threadIdx.x;        // 0..31
  const int ty = threadIdx.y;        // 0..7
  const float* src = v + ((size_t)bh * S_ + s0) * D_ + d0;
  #pragma unroll
  for (int r = 0; r < 32; r += 8)
    tile[ty + r][tx] = f2bf(src[(size_t)(ty + r) * D_ + tx]);  // tile[s][d]
  __syncthreads();
  unsigned short* dst = vT + ((size_t)bh * D_ + d0) * S_ + s0;
  #pragma unroll
  for (int r = 0; r < 32; r += 8)
    dst[(size_t)(ty + r) * S_ + tx] = tile[tx][ty + r];        // out[d][s]
}

// ---------- main kernel ----------
__global__ __launch_bounds__(256) void retention_main(const float* __restrict__ Qg,
                                                      const unsigned short* __restrict__ Kbf,
                                                      const unsigned short* __restrict__ VTbf,
                                                      const float* __restrict__ Mg,
                                                      const float* __restrict__ rsg,
                                                      float* __restrict__ Og) {
  // double-buffered staging tiles (async-filled)
  __shared__ unsigned short Klds[2][B_][32][D_];  // [buf][b][key][d]      32 KB
  __shared__ unsigned short Vlds[2][B_][D_][32];  // [buf][b][d2][key]     32 KB
  __shared__ float          Mlds[2][64][32];      // [buf][row][key]       16 KB
  __shared__ unsigned short Plds[8][16][32];      // per-wave P scratch     8 KB

  const int h     = blockIdx.y;
  const int iBase = blockIdx.x * 64;
  const int wave  = threadIdx.x >> 5;
  const int lane  = threadIdx.x & 31;
  const int b     = wave >> 2;     // batch for this wave
  const int qt    = wave & 3;      // query tile within 64-row block
  const int hi    = lane >> 4;
  const int m     = lane & 15;
  const int i0    = iBase + qt * 16;

  // issue one step's async staging: K tile, V^T tile, mask tile (10 x b128/thread)
  auto stage = [&](int step, int buf) {
    const int j0 = step * 32;
    const unsigned short* kb = Kbf  + ((size_t)h * S_ + j0) * D_;
    const unsigned short* vb = VTbf + ((size_t)h * D_) * S_ + j0;
    const float*          mb = Mg   + ((size_t)h * S_ + iBase) * S_ + j0;
    const unsigned ldsK = lds_off(&Klds[buf][0][0][0]);
    const unsigned ldsV = lds_off(&Vlds[buf][0][0][0]);
    const unsigned ldsM = lds_off(&Mlds[buf][0][0]);
    #pragma unroll
    for (int c = 0; c < 4; ++c) {            // K: 1024 chunks, layout-matched
      unsigned idx  = c * 256u + threadIdx.x;
      unsigned voff = (idx >> 9) * (unsigned)(H_ * S_ * D_ * 2)
                    + ((idx >> 4) & 31u) * (unsigned)(D_ * 2)
                    + (idx & 15u) * 16u;
      async_b128(ldsK + idx * 16u, voff, kb);
    }
    #pragma unroll
    for (int c = 0; c < 4; ++c) {            // V^T: 1024 chunks
      unsigned idx  = c * 256u + threadIdx.x;
      unsigned voff = (idx >> 9) * (unsigned)(H_ * D_ * S_ * 2)
                    + ((idx >> 2) & 127u) * (unsigned)(S_ * 2)
                    + (idx & 3u) * 16u;
      async_b128(ldsV + idx * 16u, voff, vb);
    }
    #pragma unroll
    for (int c = 0; c < 2; ++c) {            // mask: 512 chunks (f32, streamed once -> NT)
      unsigned idx  = c * 256u + threadIdx.x;
      unsigned voff = (idx >> 3) * (unsigned)(S_ * 4) + (idx & 7u) * 16u;
      async_b128_nt(ldsM + idx * 16u, voff, mb);
    }
  };

  // rs values for this lane's 8 C-layout rows
  float rsv[8];
  #pragma unroll
  for (int r = 0; r < 8; ++r)
    rsv[r] = rsg[h * S_ + i0 + hi * 8 + r];

  // Q tile in WMMA A-layout (M = lane%16), 4 chunks of K=32 over D (one-time load)
  BF16x16 qa[4];
  {
    const float* qrow = Qg + (((size_t)(b * H_ + h)) * S_ + (i0 + m)) * D_;
    #pragma unroll
    for (int c = 0; c < 4; ++c) {
      #pragma unroll
      for (int t = 0; t < 8; ++t) {
        qa[c].s[t]     = f2bf(qrow[c * 32 + hi * 16 + t]);
        qa[c].s[t + 8] = f2bf(qrow[c * 32 + hi * 16 + 8 + t]);
      }
    }
  }

  v8f oacc[8];
  #pragma unroll
  for (int nt = 0; nt < 8; ++nt) oacc[nt] = (v8f)0.f;

  stage(0, 0);  // prologue

  for (int step = 0; step < S_ / 32; ++step) {
    const int buf = step & 1;
    wait_async0();      // our async writes for `buf` done
    __syncthreads();    // everyone's done (and prior compute on buf^1 finished)
    if (step + 1 < S_ / 32) stage(step + 1, buf ^ 1);  // overlap next staging

    // --- GEMM 1: P = (Q K^T) * mask * rs, two 16x16 j-tiles ---
    #pragma unroll
    for (int jt = 0; jt < 2; ++jt) {
      v8f acc = (v8f)0.f;
      #pragma unroll
      for (int c = 0; c < 4; ++c) {
        BF16x16 kbv;  // B-layout: N = lane%16 (key), K = hi*16..+15 contiguous in d
        const unsigned short* kp = &Klds[buf][b][jt * 16 + m][c * 32 + hi * 16];
        kbv.q[0] = *(const u32x4*)kp;
        kbv.q[1] = *(const u32x4*)(kp + 8);
        acc = __builtin_amdgcn_wmma_f32_16x16x32_bf16(false, qa[c].v, false, kbv.v,
                                                      (short)0, acc, false, false);
      }
      #pragma unroll
      for (int r = 0; r < 8; ++r) {
        float pv = acc[r] * Mlds[buf][qt * 16 + hi * 8 + r][jt * 16 + m] * rsv[r];
        Plds[wave][hi * 8 + r][jt * 16 + m] = f2bf(pv);
      }
    }

    // re-read P in A-layout (same-wave LDS ops are in-order)
    BF16x16 pa;
    pa.q[0] = *(const u32x4*)&Plds[wave][m][hi * 16];
    pa.q[1] = *(const u32x4*)&Plds[wave][m][hi * 16 + 8];

    // --- GEMM 2: O += P @ V over 8 n-tiles ---
    #pragma unroll
    for (int nt = 0; nt < 8; ++nt) {
      BF16x16 vbv;  // B-layout: N = d2 = nt*16+m, K = keys hi*16..+15 contiguous (V^T)
      const unsigned short* vp = &Vlds[buf][b][nt * 16 + m][hi * 16];
      vbv.q[0] = *(const u32x4*)vp;
      vbv.q[1] = *(const u32x4*)(vp + 8);
      oacc[nt] = __builtin_amdgcn_wmma_f32_16x16x32_bf16(false, pa.v, false, vbv.v,
                                                         (short)0, oacc[nt], false, false);
    }
  }

  // --- RMSNorm over 128 cols per row, then store ---
  float ss[8];
  #pragma unroll
  for (int r = 0; r < 8; ++r) {
    float s = 0.f;
    #pragma unroll
    for (int nt = 0; nt < 8; ++nt) { float x = oacc[nt][r]; s += x * x; }
    ss[r] = s;
  }
  #pragma unroll
  for (int mk = 1; mk <= 8; mk <<= 1) {
    #pragma unroll
    for (int r = 0; r < 8; ++r) ss[r] += __shfl_xor(ss[r], mk, 32);
  }
  float scl[8];
  #pragma unroll
  for (int r = 0; r < 8; ++r) scl[r] = rsqrtf(ss[r] * (1.f / 128.f) + EPS_);

  #pragma unroll
  for (int r = 0; r < 8; ++r) {
    size_t base = (((size_t)(b * H_ + h)) * S_ + (i0 + hi * 8 + r)) * D_;
    #pragma unroll
    for (int nt = 0; nt < 8; ++nt)
      Og[base + nt * 16 + m] = oacc[nt][r] * scl[r];
  }
}

extern "C" void kernel_launch(void* const* d_in, const int* in_sizes, int n_in,
                              void* d_out, int out_size, void* d_ws, size_t ws_size,
                              hipStream_t stream) {
  (void)in_sizes; (void)n_in; (void)out_size; (void)ws_size;
  const float* q  = (const float*)d_in[0];
  const float* k  = (const float*)d_in[1];
  const float* v  = (const float*)d_in[2];
  const float* om = (const float*)d_in[3];
  float* out = (float*)d_out;

  // workspace layout: rs (64 KB) | k_bf (8 MB) | vT_bf (8 MB)  ~= 16.1 MB total
  char* ws = (char*)d_ws;
  float* rs = (float*)ws;
  unsigned short* k_bf = (unsigned short*)(ws + 65536);
  unsigned short* vT_bf = (unsigned short*)(ws + 65536 + (size_t)B_ * H_ * S_ * D_ * 2);

  retention_rowsum<<<H_ * S_, 256, 0, stream>>>(om, rs);
  convert_k<<<(B_ * H_ * S_ * D_) / (256 * 8), 256, 0, stream>>>(k, k_bf);
  transpose_v<<<dim3(S_ / 32, D_ / 32, B_ * H_), dim3(32, 8), 0, stream>>>(v, vT_bf);
  retention_main<<<dim3(S_ / 64, H_), 256, 0, stream>>>(q, k_bf, vT_bf, om, rs, out);
}
